// HardNegativeInfoNCELoss_29841432773247
// MI455X (gfx1250) — compile-verified, matrix operand order checked by reading, not run on hardware
//
#include <hip/hip_runtime.h>
#include <math.h>

// ---------- types ----------
typedef __attribute__((ext_vector_type(16))) __bf16 v16bf;
typedef __attribute__((ext_vector_type(8)))  float  v8f;
typedef unsigned uv4 __attribute__((ext_vector_type(4)));
typedef unsigned uv2 __attribute__((ext_vector_type(2)));

union FragU { uv4 q[2]; v16bf v; };

#define B_ANCH   1024
#define DIMK     256
#define M_POOL   262144
#define MTILE    256
#define TOPK     10
#define NGROUPS  2                 // N-tile groups per block (8 tiles = 128 cols each)
#define CH_STRIDE 264              // ushorts per LDS chunk row (256 + 8 pad -> conflict-free b128)
#define INV_T    (1.0f/0.07f)
#define NEG_INF  (-3.402823466e38f)

#if __has_builtin(__builtin_amdgcn_sched_barrier)
#define SCHED_FENCE() __builtin_amdgcn_sched_barrier(0)
#else
#define SCHED_FENCE() asm volatile("" ::: "memory")
#endif

__device__ __forceinline__ unsigned pack_bf16_2(float lo, float hi) {
  unsigned a = __float_as_uint(lo), b = __float_as_uint(hi);
  a = (a + 0x7FFFu + ((a >> 16) & 1u)) >> 16;     // RNE
  b = (b + 0x7FFFu + ((b >> 16) & 1u)) >> 16;
  return a | (b << 16);
}

// B-fragment loader; idx = kk*... flattened (kk = idx>>3, nt = idx&7). Constant-folds
// to immediate LDS offsets after full unroll.
__device__ __forceinline__ void load_bfrag(FragU& d, const unsigned short* bbase, int idx) {
  const unsigned short* p =
      bbase + (size_t)((idx & 7) * 16) * CH_STRIDE + (idx >> 3) * 32;
  d.q[0] = *(const uv4*)(p);
  d.q[1] = *(const uv4*)(p + 16);
}

// ---------- kernel 1: normalize anchors/positives, bf16 anchors + pos_sim/T ----------
__global__ __launch_bounds__(256)
void prep_kernel(const float* __restrict__ anchor, const float* __restrict__ positive,
                 unsigned short* __restrict__ abf, float* __restrict__ pos_sim) {
  int tid = threadIdx.x, lane = tid & 31, w = tid >> 5;
  int b = blockIdx.x * 8 + w;                      // one anchor per wave
  const float4* a4 = (const float4*)(anchor   + (size_t)b * DIMK + lane * 8);
  const float4* p4 = (const float4*)(positive + (size_t)b * DIMK + lane * 8);
  float4 a0 = a4[0], a1 = a4[1], p0 = p4[0], p1 = p4[1];
  float ssa = a0.x*a0.x + a0.y*a0.y + a0.z*a0.z + a0.w*a0.w
            + a1.x*a1.x + a1.y*a1.y + a1.z*a1.z + a1.w*a1.w;
  float ssp = p0.x*p0.x + p0.y*p0.y + p0.z*p0.z + p0.w*p0.w
            + p1.x*p1.x + p1.y*p1.y + p1.z*p1.z + p1.w*p1.w;
  float dp  = a0.x*p0.x + a0.y*p0.y + a0.z*p0.z + a0.w*p0.w
            + a1.x*p1.x + a1.y*p1.y + a1.z*p1.z + a1.w*p1.w;
#pragma unroll
  for (int off = 16; off > 0; off >>= 1) {
    ssa += __shfl_xor(ssa, off, 32);
    ssp += __shfl_xor(ssp, off, 32);
    dp  += __shfl_xor(dp,  off, 32);
  }
  float inva = 1.0f / fmaxf(sqrtf(ssa), 1e-12f);
  float invp = 1.0f / fmaxf(sqrtf(ssp), 1e-12f);
  uv4 o;
  o.x = pack_bf16_2(a0.x*inva, a0.y*inva);
  o.y = pack_bf16_2(a0.z*inva, a0.w*inva);
  o.z = pack_bf16_2(a1.x*inva, a1.y*inva);
  o.w = pack_bf16_2(a1.z*inva, a1.w*inva);
  *(uv4*)(abf + (size_t)b * DIMK + lane * 8) = o;  // 16B aligned
  if (lane == 0) pos_sim[b] = dp * inva * invp * INV_T;
}

// ---------- kernel 2: normalize pool chunk -> LDS bf16, WMMA GEMM, per-group top-10 ----------
__global__ __launch_bounds__(256)
void gemm_topk_kernel(const float* __restrict__ npool, const unsigned short* __restrict__ abf,
                      float* __restrict__ partial, int numBlk) {
  extern __shared__ unsigned short s_chunk[];      // MTILE x CH_STRIDE bf16 (132 KB)
  int blk = blockIdx.x, tid = threadIdx.x, lane = tid & 31, w = tid >> 5;
  int r = lane & 15, half = lane >> 4;

  // Phase 1: each thread owns one pool row: sumsq pass, then normalize+convert into LDS.
  {
    const float4* s4 = (const float4*)(npool + ((size_t)blk * MTILE + tid) * DIMK);
    float ss = 0.0f;
#pragma unroll 8
    for (int i = 0; i < 64; ++i) { float4 v = s4[i]; ss += v.x*v.x + v.y*v.y + v.z*v.z + v.w*v.w; }
    float inv = 1.0f / fmaxf(sqrtf(ss), 1e-12f);
    uv2* crow = (uv2*)(s_chunk + (size_t)tid * CH_STRIDE);
#pragma unroll 8
    for (int i = 0; i < 64; ++i) {
      float4 v = s4[i];                            // re-read (L2 hot)
      uv2 o; o.x = pack_bf16_2(v.x*inv, v.y*inv); o.y = pack_bf16_2(v.z*inv, v.w*inv);
      crow[i] = o;
    }
  }
  __syncthreads();

  unsigned halfmask = half ? 0xFFFF0000u : 0x0000FFFFu;

  // Phase 2: each wave handles 8 anchor tiles; per tile, two N-groups of 8x16 cols.
  for (int im = 0; im < 8; ++im) {
    int m0 = (w * 8 + im) * 16;
    const unsigned short* arow = abf + (size_t)(m0 + r) * DIMK + half * 8;
    FragU a[8];
#pragma unroll
    for (int kk = 0; kk < 8; ++kk) {
      a[kk].q[0] = *(const uv4*)(arow + kk * 32);
      a[kk].q[1] = *(const uv4*)(arow + kk * 32 + 16);
    }
    for (int ng = 0; ng < NGROUPS; ++ng) {
      const unsigned short* bbase =
          s_chunk + (size_t)(ng * 128 + r) * CH_STRIDE + half * 8;

      v8f acc[8];
#pragma unroll
      for (int nt = 0; nt < 8; ++nt)
#pragma unroll
        for (int e = 0; e < 8; ++e) acc[nt][e] = 0.0f;

      // Flattened (kk,nt) loop, 3-deep B-fragment software pipeline.
      FragU b0, b1, b2, bn;
      load_bfrag(b0, bbase, 0);
      load_bfrag(b1, bbase, 1);
      load_bfrag(b2, bbase, 2);
#pragma unroll
      for (int idx = 0; idx < 64; ++idx) {         // kk = idx>>3, nt = idx&7
        if (idx < 61) load_bfrag(bn, bbase, idx + 3);
        acc[idx & 7] = __builtin_amdgcn_wmma_f32_16x16x32_bf16(
            false, a[idx >> 3].v, false, b0.v, (short)0, acc[idx & 7], false, false);
        b0 = b1; b1 = b2; b2 = bn;
        SCHED_FENCE();                             // keep manual pipeline order; no hoisting
      }

      // Top-10 per anchor row over this 128-col group.
      // Rows g (lanes 0-15) and g+8 (lanes 16-31) processed in parallel.
#pragma unroll
      for (int g = 0; g < 8; ++g) {
        float vals[8];
#pragma unroll
        for (int nt = 0; nt < 8; ++nt) vals[nt] = acc[nt][g];
        int arowg = m0 + g + 8 * half;
        float* dst = partial + (((size_t)arowg * numBlk + blk) * NGROUPS + ng) * TOPK;
        for (int rr = 0; rr < TOPK; ++rr) {
          float lm = NEG_INF; int li = 0;
#pragma unroll
          for (int j = 0; j < 8; ++j) if (vals[j] > lm) { lm = vals[j]; li = j; }
          float gm = lm;
#pragma unroll
          for (int off = 8; off > 0; off >>= 1) gm = fmaxf(gm, __shfl_xor(gm, off, 32));
          unsigned long long bal = __ballot(lm == gm) & (unsigned long long)halfmask;
          int src = (int)__ffsll(bal) - 1;
          bool win = (lane == src);
#pragma unroll
          for (int j = 0; j < 8; ++j) if (win && j == li) vals[j] = NEG_INF;
          if (lane == (half << 4)) dst[rr] = gm * INV_T;
        }
      }
    }
  }
}

// ---------- kernel 3: per-anchor exact merge of partials + 11-way log-softmax ----------
__global__ __launch_bounds__(256)
void finalize_kernel(const float* __restrict__ partial, const float* __restrict__ pos_sim,
                     float* __restrict__ lossbuf, int nParts) {
  __shared__ float lists[256][11];                 // per-thread sorted top-10 + sentinel
  __shared__ float red[256];
  __shared__ int   redi[256];
  __shared__ float topv[TOPK];
  int b = blockIdx.x, t = threadIdx.x;
  const float* base = partial + (size_t)b * nParts;

  // Per-thread sorted top-10 (registers) over strided slice.
  float t10[10];
#pragma unroll
  for (int i = 0; i < 10; ++i) t10[i] = NEG_INF;
  for (int j = t; j < nParts; j += 256) {
    float v = base[j];
    if (v > t10[9]) {
      t10[9] = v;                                  // one backward bubble pass re-sorts
#pragma unroll
      for (int i = 9; i > 0; --i) {
        float hi = fmaxf(t10[i - 1], t10[i]);
        float lo = fminf(t10[i - 1], t10[i]);
        t10[i - 1] = hi; t10[i] = lo;
      }
    }
  }
#pragma unroll
  for (int i = 0; i < 10; ++i) lists[t][i] = t10[i];
  lists[t][10] = NEG_INF;                          // sentinel
  __syncthreads();

  // 10 rounds of block-wide argmax over per-thread sorted heads (exact merge).
  int ptr = 0;
  for (int rr = 0; rr < TOPK; ++rr) {
    red[t] = lists[t][ptr]; redi[t] = t;
    __syncthreads();
    for (int s = 128; s > 0; s >>= 1) {
      if (t < s && red[t + s] > red[t]) { red[t] = red[t + s]; redi[t] = redi[t + s]; }
      __syncthreads();
    }
    if (t == redi[0]) ptr++;
    if (t == 0) topv[rr] = red[0];
    __syncthreads();
  }

  if (t == 0) {
    float p0 = pos_sim[b];
    float m = p0;
#pragma unroll
    for (int i = 0; i < TOPK; ++i) m = fmaxf(m, topv[i]);
    float s = __expf(p0 - m);
#pragma unroll
    for (int i = 0; i < TOPK; ++i) s += __expf(topv[i] - m);
    lossbuf[b] = -(p0 - m - __logf(s));
  }
}

// ---------- kernel 4: mean over anchors ----------
__global__ __launch_bounds__(256)
void reduce_kernel(const float* __restrict__ lossbuf, float* __restrict__ out) {
  __shared__ float s[256];
  int t = threadIdx.x;
  float v = 0.0f;
  for (int j = t; j < B_ANCH; j += 256) v += lossbuf[j];
  s[t] = v; __syncthreads();
  for (int k = 128; k > 0; k >>= 1) { if (t < k) s[t] += s[t + k]; __syncthreads(); }
  if (t == 0) out[0] = s[0] / (float)B_ANCH;
}

// ---------- launch ----------
extern "C" void kernel_launch(void* const* d_in, const int* in_sizes, int n_in,
                              void* d_out, int out_size, void* d_ws, size_t ws_size,
                              hipStream_t stream) {
  (void)in_sizes; (void)n_in; (void)out_size; (void)ws_size;
  const float* anchor   = (const float*)d_in[0];
  const float* positive = (const float*)d_in[1];
  const float* npool    = (const float*)d_in[2];
  float* out = (float*)d_out;
  char* ws = (char*)d_ws;
  // workspace layout
  unsigned short* abf = (unsigned short*)ws;                 // 1024*256*2   = 512 KB
  float* pos_sim = (float*)(ws + 524288);                    // 1024*4       = 4 KB
  float* lossbuf = (float*)(ws + 524288 + 4096);             // 1024*4       = 4 KB
  float* partial = (float*)(ws + 524288 + 8192);             // 1024*1024*2*10*4 = 80 MB
  const int numBlk = M_POOL / MTILE;                         // 1024
  const int nParts = numBlk * NGROUPS * TOPK;                // 20480 per anchor
  size_t smem = (size_t)MTILE * CH_STRIDE * sizeof(unsigned short);  // 132 KB dynamic LDS

  hipFuncSetAttribute((const void*)gemm_topk_kernel,
                      hipFuncAttributeMaxDynamicSharedMemorySize, (int)smem);

  prep_kernel<<<B_ANCH / 8, 256, 0, stream>>>(anchor, positive, abf, pos_sim);
  gemm_topk_kernel<<<numBlk, 256, smem, stream>>>(npool, abf, partial, numBlk);
  finalize_kernel<<<B_ANCH, 256, 0, stream>>>(partial, pos_sim, lossbuf, nParts);
  reduce_kernel<<<1, 256, 0, stream>>>(lossbuf, out);
}
